// HSTU_38585986187606
// MI455X (gfx1250) — matrix operand order repaired and checked by reading, not compiled, not run on hardware
//
#include <hip/hip_runtime.h>
#include <hip/hip_bf16.h>
#include <math.h>

// ---------------------------------------------------------------------------
// HSTU forward for MI455X (gfx1250, wave32).  bf16 WMMA (v_wmma_f32_16x16x32_bf16)
// for the projection GEMM and both attention GEMMs, f32 accumulation everywhere.
// ---------------------------------------------------------------------------

typedef __attribute__((ext_vector_type(16))) __bf16 v16bf;
typedef __attribute__((ext_vector_type(8)))  float  v8f;

#define B_    8
#define S_    512
#define S2_   1024
#define D_    512
#define H_    8
#define A_    64
#define L_    64
#define PROJ_ 2048
#define NBLK_ 4
#define EPS_  1e-6f

static __device__ __forceinline__ v8f wmma_bf16(v16bf a, v16bf b, v8f c) {
  // 8 args: (neg_a, A, neg_b, B, c_mod, C, reuse_a, reuse_b)
  return __builtin_amdgcn_wmma_f32_16x16x32_bf16(false, a, false, b, (short)0, c,
                                                 false, false);
}

static __device__ __forceinline__ v8f zero8() {
  v8f z = {0.f, 0.f, 0.f, 0.f, 0.f, 0.f, 0.f, 0.f};
  return z;
}

// A-fragment (16x32 bf16): lane&15 = row M; lanes 0-15 hold K 0-15,
// lanes 16-31 hold K 16-31.  src row-major with leading dim ld.
static __device__ __forceinline__ v16bf load_a_frag(const __bf16* src, int ld,
                                                    int rowBase, int kBase, int lane) {
  int m    = rowBase + (lane & 15);
  int half = lane >> 4;
  v16bf a;
#pragma unroll
  for (int e = 0; e < 16; ++e) {
    int k = kBase + half * 16 + e;
    a[e] = src[(size_t)m * ld + k];
  }
  return a;
}

// B-fragment (32x16 bf16): lane&15 = col N; lanes 0-15 hold K 0-15, lanes 16-31
// hold K 16-31.  Source stored as B-transposed row-major: element (k,n) = bt[n*ld+k],
// so the packed K pairs are contiguous.
static __device__ __forceinline__ v16bf load_b_frag(const __bf16* bt, int ld,
                                                    int colBase, int kBase, int lane) {
  int n    = colBase + (lane & 15);
  int half = lane >> 4;
  v16bf b;
#pragma unroll
  for (int e = 0; e < 16; ++e) {
    int k = kBase + half * 16 + e;
    b[e] = bt[(size_t)n * ld + k];
  }
  return b;
}

// ---------------------------------------------------------------------------
// 1) Interleave item / rating embeddings -> x [B, 2S, D]  (x lives in d_out)
// ---------------------------------------------------------------------------
__global__ void k_build_x(const float* __restrict__ pe, const int* __restrict__ ratings,
                          const float* __restrict__ remb, float* __restrict__ x) {
  int idx = blockIdx.x * blockDim.x + threadIdx.x;
  if (idx >= B_ * S2_ * D_) return;
  int d = idx % D_;
  int t = (idx / D_) % S2_;
  int b = idx / (D_ * S2_);
  int s = t >> 1;
  float v;
  if ((t & 1) == 0) {
    v = pe[((size_t)b * S_ + s) * D_ + d];
  } else {
    int r = ratings[b * S_ + s];
    v = remb[(size_t)r * D_ + d];
  }
  x[idx] = v;
}

// ---------------------------------------------------------------------------
// 2) Convert uvqk [blk, D, PROJ] f32 -> wT [blk, PROJ, D] bf16 (B-transposed)
// ---------------------------------------------------------------------------
__global__ void k_cvt_w(const float* __restrict__ uvqk, __bf16* __restrict__ wT) {
  int idx = blockIdx.x * blockDim.x + threadIdx.x;
  if (idx >= NBLK_ * D_ * PROJ_) return;
  int n  = idx % PROJ_;
  int kk = (idx / PROJ_) % D_;
  int i  = idx / (PROJ_ * D_);
  wT[((size_t)i * PROJ_ + n) * D_ + kk] = (__bf16)uvqk[idx];
}

// ---------------------------------------------------------------------------
// 3) LayerNorm1: one wave32 per 512-wide row, output bf16
// ---------------------------------------------------------------------------
__global__ void k_ln1(const float* __restrict__ x, const float* __restrict__ w,
                      const float* __restrict__ bias, __bf16* __restrict__ out) {
  int wave = threadIdx.x >> 5, lane = threadIdx.x & 31;
  int row  = blockIdx.x * 4 + wave;
  const float* xr = x + (size_t)row * D_;
  float vals[16];
  float sum = 0.f, sq = 0.f;
#pragma unroll
  for (int j = 0; j < 16; ++j) {
    float v = xr[lane + j * 32];
    vals[j] = v; sum += v; sq += v * v;
  }
#pragma unroll
  for (int off = 16; off >= 1; off >>= 1) {
    sum += __shfl_xor(sum, off, 32);
    sq  += __shfl_xor(sq,  off, 32);
  }
  float mu   = sum * (1.0f / D_);
  float var  = sq * (1.0f / D_) - mu * mu;
  float rstd = rsqrtf(var + EPS_);
#pragma unroll
  for (int j = 0; j < 16; ++j) {
    int d = lane + j * 32;
    out[(size_t)row * D_ + d] = (__bf16)((vals[j] - mu) * rstd * w[d] + bias[d]);
  }
}

// ---------------------------------------------------------------------------
// 4) Projection GEMM: proj = normed(bf16) @ uvqk, scattered into u/q/k/vT layouts.
//    64x64 output tile per block, 4 waves, each wave = 16 rows x 4 WMMA tiles.
// ---------------------------------------------------------------------------
__global__ void k_proj(const __bf16* __restrict__ nrm, const __bf16* __restrict__ wT,
                       float* __restrict__ u, __bf16* __restrict__ qout,
                       __bf16* __restrict__ kout, __bf16* __restrict__ vtout) {
  int lane = threadIdx.x & 31, wave = threadIdx.x >> 5;
  int rowBase = blockIdx.y * 64 + wave * 16;   // token rows in [0, B*2S)
  int colBase = blockIdx.x * 64;               // proj cols  in [0, 2048)

  v8f acc[4];
#pragma unroll
  for (int t = 0; t < 4; ++t) acc[t] = zero8();

  for (int kc = 0; kc < D_ / 32; ++kc) {
    v16bf a = load_a_frag(nrm, D_, rowBase, kc * 32, lane);
#pragma unroll
    for (int t = 0; t < 4; ++t) {
      v16bf b = load_b_frag(wT, D_, colBase + t * 16, kc * 32, lane);
      acc[t] = wmma_bf16(a, b, acc[t]);
    }
  }

  int half = lane >> 4, nlo = lane & 15;
#pragma unroll
  for (int t = 0; t < 4; ++t) {
    int n = colBase + t * 16 + nlo;
#pragma unroll
    for (int e = 0; e < 8; ++e) {
      int gr  = rowBase + e + half * 8;       // global token row
      int b   = gr / S2_;
      int tok = gr % S2_;
      float val = acc[t][e];
      if (n < 512) {                                   // u, keep f32
        u[(size_t)gr * 512 + n] = val;
      } else if (n < 1024) {                           // v -> vT [b,h,l,tok]
        int c = n - 512;
        vtout[(((size_t)(b * H_ + (c >> 6))) * L_ + (c & 63)) * S2_ + tok] = (__bf16)val;
      } else if (n < 1536) {                           // q -> [b,h,tok,a]
        int c = n - 1024;
        qout[(((size_t)(b * H_ + (c >> 6))) * S2_ + tok) * A_ + (c & 63)] = (__bf16)val;
      } else {                                         // k -> [b,h,tok,a]
        int c = n - 1536;
        kout[(((size_t)(b * H_ + (c >> 6))) * S2_ + tok) * A_ + (c & 63)] = (__bf16)val;
      }
    }
  }
}

// ---------------------------------------------------------------------------
// 5) Attention: per (b,h) head, per 64-query tile.  S = silu(QK^T)*causal, O = S@V.
//    S goes through LDS to re-layout D-fragments into A-fragments.
// ---------------------------------------------------------------------------
__global__ void k_attn(const __bf16* __restrict__ qm, const __bf16* __restrict__ km,
                       const __bf16* __restrict__ vm, float* __restrict__ attn) {
  __shared__ __bf16 sS[4][16][72];   // per-wave 16x64 strip, padded
  int lane = threadIdx.x & 31, wave = threadIdx.x >> 5;
  int qt = blockIdx.x;
  int bh = blockIdx.y;               // b*H + h
  int b = bh >> 3, h = bh & 7;
  const __bf16* qp = qm + (size_t)bh * S2_ * A_;
  const __bf16* kp = km + (size_t)bh * S2_ * A_;
  const __bf16* vp = vm + (size_t)bh * L_ * S2_;

  int qRowBase = qt * 64 + wave * 16;
  int half = lane >> 4, nlo = lane & 15;

  v16bf aq[2];
  aq[0] = load_a_frag(qp, A_, qRowBase, 0, lane);
  aq[1] = load_a_frag(qp, A_, qRowBase, 32, lane);

  v8f accO[4];
#pragma unroll
  for (int t = 0; t < 4; ++t) accO[t] = zero8();

  for (int jt = 0; jt <= qt; ++jt) {
    int keyBase = jt * 64;
    // ---- S = silu(Q K^T) * causal mask, per 16x16 subtile ----
#pragma unroll
    for (int t = 0; t < 4; ++t) {
      v8f s = zero8();
#pragma unroll
      for (int kc = 0; kc < 2; ++kc) {
        v16bf bk = load_b_frag(kp, A_, keyBase + t * 16, kc * 32, lane);
        s = wmma_bf16(aq[kc], bk, s);
      }
      int keyCol = keyBase + t * 16 + nlo;
#pragma unroll
      for (int e = 0; e < 8; ++e) {
        int m    = e + half * 8;
        int qrow = qRowBase + m;
        float v  = s[e];
        v = (keyCol <= qrow) ? v / (1.0f + __expf(-v)) : 0.0f;   // silu * mask
        sS[wave][m][t * 16 + nlo] = (__bf16)v;
      }
    }
    __syncthreads();

    // ---- reload S as A-fragments (2 K-chunks over 64 keys) ----
    v16bf aS[2];
#pragma unroll
    for (int kc = 0; kc < 2; ++kc) {
      int m = nlo;
#pragma unroll
      for (int e = 0; e < 16; ++e) {
        int kk = kc * 32 + half * 16 + e;
        aS[kc][e] = sS[wave][m][kk];
      }
    }

    // ---- O += S @ V  (V stored transposed: vp[l*S2 + tok]) ----
#pragma unroll
    for (int t = 0; t < 4; ++t) {
#pragma unroll
      for (int kc = 0; kc < 2; ++kc) {
        v16bf bv;
        int l = t * 16 + nlo;
#pragma unroll
        for (int e = 0; e < 16; ++e) {
          int kI = kc * 32 + half * 16 + e;
          bv[e] = vp[(size_t)l * S2_ + keyBase + kI];
        }
        accO[t] = wmma_bf16(aS[kc], bv, accO[t]);
      }
    }
    __syncthreads();
  }

  // ---- store O -> attn [b, tok, h*64 + l] f32 ----
#pragma unroll
  for (int t = 0; t < 4; ++t) {
#pragma unroll
    for (int e = 0; e < 8; ++e) {
      int m   = e + half * 8;
      int tok = qRowBase + m;
      int l   = t * 16 + nlo;
      attn[((size_t)(b * S2_ + tok)) * 512 + h * 64 + l] = accO[t][e];
    }
  }
}

// ---------------------------------------------------------------------------
// 6) x += layernorm(attn)*ln2_w + ln2_b, gated by u.  One wave32 per row.
// ---------------------------------------------------------------------------
__global__ void k_upd(const float* __restrict__ attn, const float* __restrict__ u,
                      const float* __restrict__ w2, const float* __restrict__ b2,
                      float* __restrict__ x) {
  int wave = threadIdx.x >> 5, lane = threadIdx.x & 31;
  int row  = blockIdx.x * 4 + wave;
  const float* ar = attn + (size_t)row * D_;
  float vals[16];
  float sum = 0.f, sq = 0.f;
#pragma unroll
  for (int j = 0; j < 16; ++j) {
    float v = ar[lane + j * 32];
    vals[j] = v; sum += v; sq += v * v;
  }
#pragma unroll
  for (int off = 16; off >= 1; off >>= 1) {
    sum += __shfl_xor(sum, off, 32);
    sq  += __shfl_xor(sq,  off, 32);
  }
  float mu   = sum * (1.0f / D_);
  float var  = sq * (1.0f / D_) - mu * mu;
  float rstd = rsqrtf(var + EPS_);
#pragma unroll
  for (int j = 0; j < 16; ++j) {
    int d = lane + j * 32;
    float g = (vals[j] - mu) * rstd * w2[d] + b2[d];
    x[(size_t)row * D_ + d] += g * u[(size_t)row * D_ + d];
  }
}

// ---------------------------------------------------------------------------
extern "C" void kernel_launch(void* const* d_in, const int* in_sizes, int n_in,
                              void* d_out, int out_size, void* d_ws, size_t ws_size,
                              hipStream_t stream) {
  (void)in_sizes; (void)n_in; (void)out_size; (void)ws_size;
  // setup_inputs order:
  // 0 past_lengths (unused), 1 past_ids (unused), 2 past_embeddings,
  // 3 timestamps (unused), 4 ratings, 5 rating_emb, 6 uvqk,
  // 7 ln1_w, 8 ln1_b, 9 ln2_w, 10 ln2_b
  const float* past_emb   = (const float*)d_in[2];
  const int*   ratings    = (const int*)d_in[4];
  const float* rating_emb = (const float*)d_in[5];
  const float* uvqk       = (const float*)d_in[6];
  const float* ln1w       = (const float*)d_in[7];
  const float* ln1b       = (const float*)d_in[8];
  const float* ln2w       = (const float*)d_in[9];
  const float* ln2b       = (const float*)d_in[10];
  float* x = (float*)d_out;   // x lives in the output buffer, updated in place

  char* ws = (char*)d_ws;     // workspace layout (72 MB total)
  __bf16* nrm  = (__bf16*)(ws);                 //  8 MB  normed x, bf16
  __bf16* wT   = (__bf16*)(ws + (8u  << 20));   //  8 MB  uvqk^T bf16, all 4 blocks
  float*  u    = (float*) (ws + (16u << 20));   // 16 MB  u gate, f32
  __bf16* qb   = (__bf16*)(ws + (32u << 20));   //  8 MB  q [b,h,tok,a]
  __bf16* kb   = (__bf16*)(ws + (40u << 20));   //  8 MB  k [b,h,tok,a]
  __bf16* vT   = (__bf16*)(ws + (48u << 20));   //  8 MB  v^T [b,h,l,tok]
  float*  attn = (float*) (ws + (56u << 20));   // 16 MB  attention output

  int n1 = B_ * S2_ * D_;
  k_build_x<<<(n1 + 255) / 256, 256, 0, stream>>>(past_emb, ratings, rating_emb, x);
  int n2 = NBLK_ * D_ * PROJ_;
  k_cvt_w<<<(n2 + 255) / 256, 256, 0, stream>>>(uvqk, wT);

  for (int blk = 0; blk < NBLK_; ++blk) {
    k_ln1<<<B_ * S2_ / 4, 128, 0, stream>>>(x, ln1w + blk * D_, ln1b + blk * D_, nrm);
    dim3 gp(PROJ_ / 64, (B_ * S2_) / 64);
    k_proj<<<gp, 128, 0, stream>>>(nrm, wT + (size_t)blk * PROJ_ * D_, u, qb, kb, vT);
    dim3 ga(S2_ / 64, B_ * H_);
    k_attn<<<ga, 128, 0, stream>>>(qb, kb, vT, attn);
    k_upd<<<B_ * S2_ / 4, 128, 0, stream>>>(attn, u, ln2w + blk * D_, ln2b + blk * D_, x);
  }
}
